// MixtralBLockSparseTop2MLP_HQQ_1726576855473
// MI455X (gfx1250) — matrix operand order repaired and checked by reading, not compiled, not run on hardware
//
#include <hip/hip_runtime.h>
#include <hip/hip_bf16.h>

// Problem constants (match the reference)
constexpr int HID = 4096;
constexpr int INTR = 14336;
constexpr int GS  = 64;
constexpr int TOK = 4096;

typedef __attribute__((ext_vector_type(16))) _Float16 v16h;
typedef __attribute__((ext_vector_type(8)))  _Float16 v8h;
typedef __attribute__((ext_vector_type(8)))  float    v8f;

// Token-stationary tiling: the 705 MB of int32 weight codes stream only
// TOK/BM = 16 times; the A panel per block-row lives in the 192 MB L2.
constexpr int BM   = 256;        // tokens per block
constexpr int BN   = 64;         // output cols per block
constexpr int KBLK = 64;         // K-elements staged in LDS per barrier
constexpr int LDSS = KBLK + 8;   // +8 halves pad (144B row stride, 16B aligned)
constexpr int NTHREADS = 512;    // 16 waves: 4 (M) x 4 (N)

// ---------------------------------------------------------------------------
__device__ __forceinline__ v8f wmma_f16(v16h a, v16h b, v8f c) {
  return __builtin_amdgcn_wmma_f32_16x16x32_f16(false, a, false, b, (short)0, c,
                                                false, false);
}

// A fragment (16x32, f16). ISA 7.12.2: lane holds row M = lane&15.
__device__ __forceinline__ v16h load_a_frag(const _Float16* As, int m_base, int lane) {
  const _Float16* p = As + (m_base + (lane & 15)) * LDSS;
  int koff = (lane >= 16) ? 8 : 0;
  union { v16h v; v8h h[2]; } u;
  u.h[0] = *(const v8h*)(p + koff);
  u.h[1] = *(const v8h*)(p + 16 + koff);
  return u.v;
}

// B fragment (32x16, f16), LDS layout [N][K]. Lane holds col N = lane&15.
__device__ __forceinline__ v16h load_b_frag(const _Float16* Bs, int n_base, int lane) {
  const _Float16* p = Bs + (n_base + (lane & 15)) * LDSS + ((lane >= 16) ? 8 : 0);
  union { v16h v; v8h h[2]; } u;
  u.h[0] = *(const v8h*)(p);
  u.h[1] = *(const v8h*)(p + 16);
  return u.v;
}

// Fast SiLU: x * sigmoid(x) with v_rcp_f32 (4-bit weights don't need IEEE div)
__device__ __forceinline__ float fast_silu(float x) {
  return x * __builtin_amdgcn_rcpf(1.0f + __expf(-x));
}

// ---------------------------------------------------------------------------
// Kernel 1: h[t,i] = silu(X·W1ᵀ) * (X·W3ᵀ), dequant fused, f16 out to ws.
// 256x64 block tile, 16 waves (4M x 4N), K-stage 64, double-buffered LDS:
// one barrier per stage, stage k+1 global loads overlap stage k WMMA.
// All fragments of a k-chunk are loaded before the WMMA chain so the
// scheduler can batch ds_load_b128s and use graduated s_wait_dscnt.
// ---------------------------------------------------------------------------
__global__ __launch_bounds__(NTHREADS)
void gate_up_silu_kernel(const float* __restrict__ X,
                         const int*   __restrict__ w1q,
                         const float* __restrict__ w1s,
                         const float* __restrict__ w1z,
                         const int*   __restrict__ w3q,
                         const float* __restrict__ w3s,
                         const float* __restrict__ w3z,
                         _Float16*    __restrict__ H) {
  __shared__ __align__(16) _Float16 As[2][BM * LDSS];   // 2 x 36.9 KB
  __shared__ __align__(16) _Float16 Bg[2][BN * LDSS];   // 2 x  9.2 KB
  __shared__ __align__(16) _Float16 Bu[2][BN * LDSS];   // 2 x  9.2 KB

  const int tid  = threadIdx.x;
  const int lane = tid & 31;
  const int wave = tid >> 5;
  const int wm   = wave >> 2;        // 0..3 : 64-token slab
  const int wn   = wave & 3;         // 0..3 : 16-col slab

  const int i0 = blockIdx.x * BN;    // intermediate dim (N)
  const int t0 = blockIdx.y * BM;    // token dim (M)

  // Cooperative-load mappings (512 threads)
  const int ar = tid >> 1;           // A: row 0..255, 32 floats
  const int ac = (tid & 1) * 32;
  const int wr = tid >> 3;           // W: row 0..63, 8 codes (one GS group)
  const int wc = (tid & 7) * 8;

  const float* xsrc  = X   + (size_t)(t0 + ar) * HID + ac;
  const int*   q1src = w1q + (size_t)(i0 + wr) * HID + wc;
  const int*   q3src = w3q + (size_t)(i0 + wr) * HID + wc;
  const float* s1src = w1s + (size_t)(i0 + wr) * (HID / GS);
  const float* z1src = w1z + (size_t)(i0 + wr) * (HID / GS);
  const float* s3src = w3s + (size_t)(i0 + wr) * (HID / GS);
  const float* z3src = w3z + (size_t)(i0 + wr) * (HID / GS);

  // Stage registers (global loads land here; converted at commit time)
  float4 xa[8];
  int    q1v[8], q3v[8];
  float  s1, z1, s3, z3;

  auto fetch = [&](int k0) {
#pragma unroll
    for (int c = 0; c < 8; ++c) xa[c] = *(const float4*)(xsrc + k0 + c * 4);
#pragma unroll
    for (int j = 0; j < 8; ++j) { q1v[j] = q1src[k0 + j]; q3v[j] = q3src[k0 + j]; }
    const int gi = k0 >> 6;
    s1 = s1src[gi]; z1 = z1src[gi]; s3 = s3src[gi]; z3 = z3src[gi];
  };

  auto commit = [&](int b) {
    _Float16* adst = &As[b][ar * LDSS + ac];
#pragma unroll
    for (int c = 0; c < 4; ++c) {
      float4 x0 = xa[2 * c], x1 = xa[2 * c + 1];
      v8h h;
      h[0] = (_Float16)x0.x; h[1] = (_Float16)x0.y;
      h[2] = (_Float16)x0.z; h[3] = (_Float16)x0.w;
      h[4] = (_Float16)x1.x; h[5] = (_Float16)x1.y;
      h[6] = (_Float16)x1.z; h[7] = (_Float16)x1.w;
      *(v8h*)(adst + c * 8) = h;
    }
    v8h b1, b3;
#pragma unroll
    for (int j = 0; j < 8; ++j) {
      b1[j] = (_Float16)(((float)q1v[j] - z1) * s1);
      b3[j] = (_Float16)(((float)q3v[j] - z3) * s3);
    }
    *(v8h*)&Bg[b][wr * LDSS + wc] = b1;
    *(v8h*)&Bu[b][wr * LDSS + wc] = b3;
  };

  v8f cg[4] = {};
  v8f cu[4] = {};

  constexpr int NSTAGE = HID / KBLK;   // 64
  fetch(0);
  commit(0);
  for (int k = 0; k < NSTAGE; ++k) {
    __syncthreads();                         // buf[k&1] published; buf[k+1&1] free
    if (k + 1 < NSTAGE) fetch((k + 1) * KBLK);   // long-latency loads in flight
    const int b = k & 1;
#pragma unroll
    for (int kk = 0; kk < KBLK; kk += 32) {
      // Batch all fragment loads of this k-chunk before the WMMA chain.
      v16h a[4];
#pragma unroll
      for (int mi = 0; mi < 4; ++mi)
        a[mi] = load_a_frag(&As[b][kk], wm * 64 + mi * 16, lane);
      v16h bg = load_b_frag(&Bg[b][kk], wn * 16, lane);
      v16h bu = load_b_frag(&Bu[b][kk], wn * 16, lane);
#pragma unroll
      for (int mi = 0; mi < 4; ++mi) {
        cg[mi] = wmma_f16(a[mi], bg, cg[mi]);
        cu[mi] = wmma_f16(a[mi], bu, cu[mi]);
      }
    }
    if (k + 1 < NSTAGE) commit((k + 1) & 1);
  }

  // ---- epilogue: SwiGLU; C/D layout: M = v + 8*(lane>=16), N = lane&15
  const int mrow = (lane >= 16) ? 8 : 0;
  const int ncol = lane & 15;
#pragma unroll
  for (int mi = 0; mi < 4; ++mi)
#pragma unroll
    for (int v = 0; v < 8; ++v) {
      float g = cg[mi][v];
      float u = cu[mi][v];
      float hval = fast_silu(g) * u;
      int row = t0 + wm * 64 + mi * 16 + v + mrow;
      int col = i0 + wn * 16 + ncol;
      H[(size_t)row * INTR + col] = (_Float16)hval;
    }
}

// ---------------------------------------------------------------------------
// Kernel 2: out[t,h] = sum_i h[t,i] * dequant(W2)[h,i]. Same double-buffered
// pipeline, 8 v_wmma per stage, f32 output.
// ---------------------------------------------------------------------------
__global__ __launch_bounds__(NTHREADS)
void down_proj_kernel(const _Float16* __restrict__ H,
                      const int*      __restrict__ w2q,
                      const float*    __restrict__ w2s,
                      const float*    __restrict__ w2z,
                      float*          __restrict__ O) {
  __shared__ __align__(16) _Float16 As[2][BM * LDSS];
  __shared__ __align__(16) _Float16 Bs[2][BN * LDSS];

  const int tid  = threadIdx.x;
  const int lane = tid & 31;
  const int wave = tid >> 5;
  const int wm   = wave >> 2;
  const int wn   = wave & 3;

  const int h0 = blockIdx.x * BN;    // output hidden dim (N)
  const int t0 = blockIdx.y * BM;    // tokens (M)

  const int ar = tid >> 1;
  const int ac = (tid & 1) * 32;     // 32 halves
  const int wr = tid >> 3;
  const int wc = (tid & 7) * 8;

  const _Float16* hsrc = H   + (size_t)(t0 + ar) * INTR + ac;
  const int*      qsrc = w2q + (size_t)(h0 + wr) * INTR + wc;
  const float*    ssrc = w2s + (size_t)(h0 + wr) * (INTR / GS);
  const float*    zsrc = w2z + (size_t)(h0 + wr) * (INTR / GS);

  v8h   ha[4];
  int   qv[8];
  float s2, z2;

  auto fetch = [&](int k0) {
#pragma unroll
    for (int c = 0; c < 4; ++c) ha[c] = *(const v8h*)(hsrc + k0 + c * 8);
#pragma unroll
    for (int j = 0; j < 8; ++j) qv[j] = qsrc[k0 + j];
    const int gi = k0 >> 6;
    s2 = ssrc[gi]; z2 = zsrc[gi];
  };

  auto commit = [&](int b) {
    _Float16* adst = &As[b][ar * LDSS + ac];
#pragma unroll
    for (int c = 0; c < 4; ++c) *(v8h*)(adst + c * 8) = ha[c];
    v8h bv;
#pragma unroll
    for (int j = 0; j < 8; ++j) bv[j] = (_Float16)(((float)qv[j] - z2) * s2);
    *(v8h*)&Bs[b][wr * LDSS + wc] = bv;
  };

  v8f acc[4] = {};

  constexpr int NSTAGE = INTR / KBLK;   // 224
  fetch(0);
  commit(0);
  for (int k = 0; k < NSTAGE; ++k) {
    __syncthreads();
    if (k + 1 < NSTAGE) fetch((k + 1) * KBLK);
    const int b = k & 1;
#pragma unroll
    for (int kk = 0; kk < KBLK; kk += 32) {
      v16h a[4];
#pragma unroll
      for (int mi = 0; mi < 4; ++mi)
        a[mi] = load_a_frag(&As[b][kk], wm * 64 + mi * 16, lane);
      v16h bf = load_b_frag(&Bs[b][kk], wn * 16, lane);
#pragma unroll
      for (int mi = 0; mi < 4; ++mi)
        acc[mi] = wmma_f16(a[mi], bf, acc[mi]);
    }
    if (k + 1 < NSTAGE) commit((k + 1) & 1);
  }

  const int mrow = (lane >= 16) ? 8 : 0;
  const int ncol = lane & 15;
#pragma unroll
  for (int mi = 0; mi < 4; ++mi)
#pragma unroll
    for (int v = 0; v < 8; ++v) {
      int row = t0 + wm * 64 + mi * 16 + v + mrow;
      int col = h0 + wn * 16 + ncol;
      O[(size_t)row * HID + col] = acc[mi][v];
    }
}

// ---------------------------------------------------------------------------
extern "C" void kernel_launch(void* const* d_in, const int* in_sizes, int n_in,
                              void* d_out, int out_size, void* d_ws, size_t ws_size,
                              hipStream_t stream) {
  (void)in_sizes; (void)n_in; (void)out_size; (void)ws_size;
  const float* X   = (const float*)d_in[0];
  const int*   w1q = (const int*)  d_in[1];
  const float* w1s = (const float*)d_in[2];
  const float* w1z = (const float*)d_in[3];
  const int*   w3q = (const int*)  d_in[4];
  const float* w3s = (const float*)d_in[5];
  const float* w3z = (const float*)d_in[6];
  const int*   w2q = (const int*)  d_in[7];
  const float* w2s = (const float*)d_in[8];
  const float* w2z = (const float*)d_in[9];
  float*     out = (float*)d_out;
  _Float16*  H   = (_Float16*)d_ws;   // TOK*INTR f16 = ~117 MB intermediate

  dim3 blk(NTHREADS);
  gate_up_silu_kernel<<<dim3(INTR / BN, TOK / BM), blk, 0, stream>>>(
      X, w1q, w1s, w1z, w3q, w3s, w3z, H);
  down_proj_kernel<<<dim3(HID / BN, TOK / BM), blk, 0, stream>>>(
      H, w2q, w2s, w2z, out);
}
